// VectorQuantizer_15212774162947
// MI455X (gfx1250) — matrix operand order repaired and checked by reading, not compiled, not run on hardware
//
#include <hip/hip_runtime.h>
#include <hip/hip_bf16.h>
#include <stdint.h>

typedef __attribute__((ext_vector_type(16))) __bf16 v16bf;
typedef __attribute__((ext_vector_type(8)))  float  v8f;

#define NZ   131072   // rows of z
#define DIM  512      // embedding dim
#define KCB  1024     // codebook entries

struct alignas(16) U32x4 { unsigned int a, b, c, d; };

union AFrag {
    v16bf v;
    unsigned int u[8];
    U32x4 q[2];
};

// fp32 -> packed 2x bf16 (round-to-nearest-even)
__device__ __forceinline__ unsigned int pk_bf16(float a, float b) {
    unsigned int ua = __float_as_uint(a), ub = __float_as_uint(b);
    ua = (ua + 0x7FFFu + ((ua >> 16) & 1u)) >> 16;
    ub = (ub + 0x7FFFu + ((ub >> 16) & 1u)) >> 16;
    return ua | (ub << 16);
}

// ---------------------------------------------------------------------------
// Kernel 1: pack codebook into bf16 WMMA B-operand order.
// bcode[t][c][lane][j]  (t: 16-entry codebook tile, c: 32-wide K chunk)
// B 32x16 bf16, contiguous K order chosen here (A matches): lane n (0-15) = column n, K = 0..15 (2 per VGPR);
// lane n+16 = column n, K = 16..31.  K_global = c*32 + hi*16 + {0..15}.
// ---------------------------------------------------------------------------
__global__ void __launch_bounds__(256) vq_pack_codebook(
    const float* __restrict__ cb, unsigned int* __restrict__ bcode) {
    int w    = (blockIdx.x * blockDim.x + threadIdx.x) >> 5;  // 0..1023
    int lane = threadIdx.x & 31;
    int t = w >> 4, c = w & 15;
    int L16 = lane & 15, hi = lane >> 4;
    const float* row = cb + (size_t)(t * 16 + L16) * DIM + c * 32 + hi * 16;
    float4 f0 = ((const float4*)row)[0];
    float4 f1 = ((const float4*)row)[1];
    float4 f2 = ((const float4*)row)[2];
    float4 f3 = ((const float4*)row)[3];
    unsigned int* dst = bcode + ((size_t)(t * 16 + c) * 32 + lane) * 8;
    U32x4 q0 = { pk_bf16(f0.x, f0.y), pk_bf16(f0.z, f0.w),
                 pk_bf16(f1.x, f1.y), pk_bf16(f1.z, f1.w) };
    U32x4 q1 = { pk_bf16(f2.x, f2.y), pk_bf16(f2.z, f2.w),
                 pk_bf16(f3.x, f3.y), pk_bf16(f3.z, f3.w) };
    ((U32x4*)dst)[0] = q0;
    ((U32x4*)dst)[1] = q1;
}

// ---------------------------------------------------------------------------
// Kernel 2: codebook row squared norms (exact fp32)
// ---------------------------------------------------------------------------
__global__ void __launch_bounds__(256) vq_esq(
    const float* __restrict__ cb, float* __restrict__ esq) {
    int r = blockIdx.x * blockDim.x + threadIdx.x;
    if (r >= KCB) return;
    const float4* p = (const float4*)(cb + (size_t)r * DIM);
    float s = 0.f;
    for (int i = 0; i < DIM / 4; ++i) {
        float4 v = p[i];
        s += v.x * v.x + v.y * v.y + v.z * v.z + v.w * v.w;
    }
    esq[r] = s;
}

// ---------------------------------------------------------------------------
// Kernel 3: fused GEMM + argmin.  One wave owns 16 z-rows.
// A 16x32 bf16 layout: lane L16 = row, hi half picks a 16-wide contiguous K half:
//   VGPR j(0..7): K = hi*16 + 2j,2j+1 (same order as packed B).
// Score = e_sq[n] - 2*cross (z_sq is argmin-invariant).
// ---------------------------------------------------------------------------
__global__ void __launch_bounds__(128) vq_argmin(
    const float* __restrict__ z, const unsigned int* __restrict__ bcode,
    const float* __restrict__ esq, float* __restrict__ out_idx_f) {
    int w    = (blockIdx.x * blockDim.x + threadIdx.x) >> 5;
    int lane = threadIdx.x & 31;
    int L16 = lane & 15, hi = lane >> 4;
    long row_base = (long)w * 16;

    // --- preload & convert this wave's 16 z rows into bf16 A fragments ---
    AFrag a[16];
    const float* zr = z + (row_base + L16) * DIM;
#pragma unroll
    for (int c = 0; c < 16; ++c) {
        int k0 = c * 32 + hi * 16;
        float4 f0 = *(const float4*)(zr + k0);
        float4 f1 = *(const float4*)(zr + k0 + 4);
        float4 f2 = *(const float4*)(zr + k0 + 8);
        float4 f3 = *(const float4*)(zr + k0 + 12);
        U32x4 q0 = { pk_bf16(f0.x, f0.y), pk_bf16(f0.z, f0.w),
                     pk_bf16(f1.x, f1.y), pk_bf16(f1.z, f1.w) };
        U32x4 q1 = { pk_bf16(f2.x, f2.y), pk_bf16(f2.z, f2.w),
                     pk_bf16(f3.x, f3.y), pk_bf16(f3.z, f3.w) };
        a[c].q[0] = q0;
        a[c].q[1] = q1;
    }

    float best[8];
    int   bidx[8];
#pragma unroll
    for (int r = 0; r < 8; ++r) { best[r] = 3.4e38f; bidx[r] = 0; }

    for (int t = 0; t < KCB / 16; ++t) {
        v8f acc = {0.f, 0.f, 0.f, 0.f, 0.f, 0.f, 0.f, 0.f};
        const U32x4* bp =
            (const U32x4*)(bcode + ((size_t)(t * 16) * 32 + lane) * 8);
#pragma unroll
        for (int c = 0; c < 16; ++c) {
            AFrag b;
            b.q[0] = bp[c * 64];
            b.q[1] = bp[c * 64 + 1];
            acc = __builtin_amdgcn_wmma_f32_16x16x32_bf16(
                false, a[c].v, false, b.v, (short)0, acc, false, false);
        }
        int   n  = t * 16 + L16;
        float es = esq[n];
#pragma unroll
        for (int r = 0; r < 8; ++r) {
            float s = fmaf(-2.0f, acc[r], es);
            if (s < best[r]) { best[r] = s; bidx[r] = n; }  // strict < : first occurrence
        }
    }

    // cross-lane min+argmin within each 16-lane half (C layout: N = lane%16)
#pragma unroll
    for (int r = 0; r < 8; ++r) {
#pragma unroll
        for (int off = 8; off >= 1; off >>= 1) {
            float ov = __shfl_xor(best[r], off, 32);
            int   oi = __shfl_xor(bidx[r], off, 32);
            if (ov < best[r] || (ov == best[r] && oi < bidx[r])) {
                best[r] = ov;
                bidx[r] = oi;
            }
        }
    }
    if (L16 == 0) {
#pragma unroll
        for (int r = 0; r < 8; ++r)   // C layout: M = r + 8*hi
            out_idx_f[row_base + hi * 8 + r] = (float)bidx[r];
    }
}

// ---------------------------------------------------------------------------
// Kernel 4: gather z_q (== z_q_st numerically), deterministic loss partials.
// Each thread: 4 consecutive elements.  Block = 1024 elements = 2 rows.
// ---------------------------------------------------------------------------
__global__ void __launch_bounds__(256) vq_gather(
    const float* __restrict__ z, const float* __restrict__ cb,
    const float* __restrict__ idx_f, float* __restrict__ zq,
    float* __restrict__ partial) {
    __shared__ float sred[8];
    size_t e   = ((size_t)blockIdx.x * 256 + threadIdx.x) * 4;
    size_t row = e >> 9;
    int    col = (int)(e & (DIM - 1));
    int    idx = (int)idx_f[row];
    float4 zv = *(const float4*)(z + row * DIM + col);
    float4 qv = *(const float4*)(cb + (size_t)idx * DIM + col);
    *(float4*)(zq + row * DIM + col) = qv;
    float dx = qv.x - zv.x, dy = qv.y - zv.y;
    float dz = qv.z - zv.z, dw = qv.w - zv.w;
    float s = dx * dx + dy * dy + dz * dz + dw * dw;
#pragma unroll
    for (int off = 16; off >= 1; off >>= 1) s += __shfl_xor(s, off, 32);
    if ((threadIdx.x & 31) == 0) sred[threadIdx.x >> 5] = s;
    __syncthreads();
    if (threadIdx.x == 0) {
        float tot = 0.f;
#pragma unroll
        for (int i = 0; i < 8; ++i) tot += sred[i];
        partial[blockIdx.x] = tot;
    }
}

// ---------------------------------------------------------------------------
// Kernel 5: deterministic final reduction -> losses
// ---------------------------------------------------------------------------
__global__ void __launch_bounds__(256) vq_finalize(
    const float* __restrict__ partial, int nparts, float* __restrict__ losses) {
    __shared__ float sh[256];
    float s = 0.f;
    for (int i = threadIdx.x; i < nparts; i += 256) s += partial[i];
    sh[threadIdx.x] = s;
    __syncthreads();
    for (int st = 128; st >= 1; st >>= 1) {
        if (threadIdx.x < st) sh[threadIdx.x] += sh[threadIdx.x + st];
        __syncthreads();
    }
    if (threadIdx.x == 0) {
        float mse = sh[0] / ((float)NZ * (float)DIM);
        losses[0] = 0.25f * mse;  // commitment_loss
        losses[1] = mse;          // codebook_loss
    }
}

extern "C" void kernel_launch(void* const* d_in, const int* in_sizes, int n_in,
                              void* d_out, int out_size, void* d_ws, size_t ws_size,
                              hipStream_t stream) {
    const float* z  = (const float*)d_in[0];   // [131072, 512] f32
    const float* cb = (const float*)d_in[1];   // [1024, 512]   f32

    float* out   = (float*)d_out;
    float* zq    = out;                              // N*D : z_q_st
    float* oidx  = out + (size_t)NZ * DIM;           // N   : indices (as float)
    float* oloss = oidx + NZ;                        // 2   : losses

    unsigned char* ws     = (unsigned char*)d_ws;
    unsigned int* bcode   = (unsigned int*)ws;                     // 1,048,576 B
    float*        esq     = (float*)(ws + 1048576);                //     4,096 B
    float*        partial = (float*)(ws + 1048576 + 4096);         //   262,144 B

    const int nblocks_gather = (NZ * DIM) / 1024;   // 65536

    vq_pack_codebook<<<128, 256, 0, stream>>>(cb, bcode);
    vq_esq<<<(KCB + 255) / 256, 256, 0, stream>>>(cb, esq);
    vq_argmin<<<(NZ / 16) / 4, 128, 0, stream>>>(z, bcode, esq, oidx);
    vq_gather<<<nblocks_gather, 256, 0, stream>>>(z, cb, oidx, zq, partial);
    vq_finalize<<<1, 256, 0, stream>>>(partial, nblocks_gather, oloss);
}
